// Transformer_64690797412810
// MI455X (gfx1250) — matrix operand order repaired and checked by reading, not compiled, not run on hardware
//
#include <hip/hip_runtime.h>
#include <hip/hip_bf16.h>

// ---------------- model constants ----------------
#define BB   2
#define SS   1024
#define DD   1024
#define LL   2
#define HDD  64
#define HH   16
#define KVH  4
#define REPF (HH / KVH)
#define FF   2048
#define EE   8
#define VV   32000
#define NTOK (BB * SS)          // 2048
#define EPSF 1e-5f

typedef __attribute__((ext_vector_type(16))) __bf16 bf16x16;
typedef __attribute__((ext_vector_type(8)))  __bf16 bf16x8;
typedef __attribute__((ext_vector_type(2)))  __bf16 bf16x2;
typedef __attribute__((ext_vector_type(8)))  float  f32x8;

// Native bf16 conversion: backend emits hardware cvt on gfx1250 (RNE).
static __device__ __forceinline__ __bf16 f2bf(float f) { return (__bf16)f; }
static __device__ __forceinline__ unsigned pkbf(float x, float y) {
  bf16x2 v;
  v[0] = (__bf16)x;
  v[1] = (__bf16)y;
  return __builtin_bit_cast(unsigned, v);
}
static __device__ __forceinline__ float bf2f(__bf16 b) {
  unsigned short h = __builtin_bit_cast(unsigned short, b);
  unsigned u = ((unsigned)h) << 16;
  return __builtin_bit_cast(float, u);
}
static __device__ __forceinline__ f32x8 zero8() {
  f32x8 v;
#pragma unroll
  for (int i = 0; i < 8; ++i) v[i] = 0.f;
  return v;
}
static __device__ __forceinline__ f32x8 wmma_bf16(bf16x16 a, bf16x16 b, f32x8 c) {
  // D = A(16x32 bf16) * B(32x16 bf16) + C(16x16 f32)
  return __builtin_amdgcn_wmma_f32_16x16x32_bf16(false, a, false, b, (short)0, c,
                                                 false, false);
}
static __device__ __forceinline__ bf16x16 join16(bf16x8 lo, bf16x8 hi) {
  return __builtin_shufflevector(lo, hi, 0, 1, 2, 3, 4, 5, 6, 7, 8, 9, 10, 11, 12,
                                 13, 14, 15);
}

// ---------------- embedding ----------------
__global__ void tg_embed(const int* __restrict__ ids, const float* __restrict__ emb,
                         float* __restrict__ h) {
  size_t i = (size_t)blockIdx.x * 256 + threadIdx.x; // N*D total
  int n = (int)(i / DD);
  int d = (int)(i % DD);
  h[i] = emb[(size_t)ids[n] * DD + d];
}

// ---------------- rmsnorm: f32 row -> bf16 row ----------------
__global__ __launch_bounds__(256) void tg_rmsnorm(const float* __restrict__ x,
                                                  const float* __restrict__ w,
                                                  __bf16* __restrict__ out) {
  __shared__ float red[8];
  int n = blockIdx.x, tid = threadIdx.x;
  const float* xr = x + (size_t)n * DD;
  float4 v = *(const float4*)(xr + tid * 4);
  float ss = v.x * v.x + v.y * v.y + v.z * v.z + v.w * v.w;
#pragma unroll
  for (int m = 1; m < 32; m <<= 1) ss += __shfl_xor(ss, m, 32);
  if ((tid & 31) == 0) red[tid >> 5] = ss;
  __syncthreads();
  float tot = 0.f;
#pragma unroll
  for (int i = 0; i < 8; ++i) tot += red[i];
  float r = rsqrtf(tot / (float)DD + EPSF);
  float4 wv = *(const float4*)(w + tid * 4);
  uint2 packed;
  packed.x = pkbf(v.x * r * wv.x, v.y * r * wv.y);
  packed.y = pkbf(v.z * r * wv.z, v.w * r * wv.w);
  *(uint2*)(out + (size_t)n * DD + tid * 4) = packed;
}

// ---------------- generic WMMA GEMM (software-pipelined) ----------------
// C(MxN) = A(MxK bf16, optional row-gather) * W(KxN f32 -> bf16)
#define BM 128
#define BN 128
#define BK 32
enum { MODE_STORE = 0, MODE_ADD = 1, MODE_SILU_MUL = 2, MODE_SCATTER = 3 };

__global__ __launch_bounds__(256) void tg_gemm(
    const __bf16* __restrict__ A, int lda, const float* __restrict__ W, int ldb,
    int M, int Kd, const int* __restrict__ amap, const int* __restrict__ cntPtr,
    int cntIdx, int mode, int ldo, float* __restrict__ outf,
    __bf16* __restrict__ outbf, const float* __restrict__ aux,
    const int* __restrict__ omap, const float* __restrict__ oscale) {
  __shared__ __bf16 As[BM][BK];   // row-major M x K
  __shared__ __bf16 Bst[BN][BK];  // transposed [n][k]: fragment = 32B contiguous
  int tid = threadIdx.x;
  int wave = tid >> 5, lane = tid & 31;
  int rlo = lane & 15, hf = lane >> 4;
  int r0 = blockIdx.y * BM, c0 = blockIdx.x * BN;
  int valid = cntPtr ? cntPtr[cntIdx] : M;
  if (r0 >= valid) return;

  f32x8 acc[8];
#pragma unroll
  for (int t = 0; t < 8; ++t) acc[t] = zero8();

  // staging thread mapping
  int sRow = tid >> 1;
  int sHoff = (tid & 1) * 16;
  int sKk = tid >> 4;        // 0..15 (and +16)
  int sNn = (tid & 15) * 8;  // 0..120

  union AU { bf16x16 v; __bf16 e[16]; };
  AU aCur;
  float4 bCur[4];

  auto stage_load = [&](int k0) {
    int gr = r0 + sRow;
    if (gr < valid) {
      int ar = amap ? amap[gr] : gr;
      aCur.v = *(const bf16x16*)(A + (size_t)ar * lda + k0 + sHoff);
    } else {
#pragma unroll
      for (int j = 0; j < 16; ++j)
        aCur.e[j] = __builtin_bit_cast(__bf16, (unsigned short)0);
    }
    const float* s0 = W + (size_t)(k0 + sKk) * ldb + c0 + sNn;
    bCur[0] = *(const float4*)s0;
    bCur[1] = *(const float4*)(s0 + 4);
    const float* s1 = W + (size_t)(k0 + sKk + 16) * ldb + c0 + sNn;
    bCur[2] = *(const float4*)s1;
    bCur[3] = *(const float4*)(s1 + 4);
  };
  auto stage_store = [&]() {
    *(bf16x16*)&As[sRow][sHoff] = aCur.v;
    Bst[sNn + 0][sKk] = f2bf(bCur[0].x);
    Bst[sNn + 1][sKk] = f2bf(bCur[0].y);
    Bst[sNn + 2][sKk] = f2bf(bCur[0].z);
    Bst[sNn + 3][sKk] = f2bf(bCur[0].w);
    Bst[sNn + 4][sKk] = f2bf(bCur[1].x);
    Bst[sNn + 5][sKk] = f2bf(bCur[1].y);
    Bst[sNn + 6][sKk] = f2bf(bCur[1].z);
    Bst[sNn + 7][sKk] = f2bf(bCur[1].w);
    Bst[sNn + 0][sKk + 16] = f2bf(bCur[2].x);
    Bst[sNn + 1][sKk + 16] = f2bf(bCur[2].y);
    Bst[sNn + 2][sKk + 16] = f2bf(bCur[2].z);
    Bst[sNn + 3][sKk + 16] = f2bf(bCur[2].w);
    Bst[sNn + 4][sKk + 16] = f2bf(bCur[3].x);
    Bst[sNn + 5][sKk + 16] = f2bf(bCur[3].y);
    Bst[sNn + 6][sKk + 16] = f2bf(bCur[3].z);
    Bst[sNn + 7][sKk + 16] = f2bf(bCur[3].w);
  };

  stage_load(0);
  int arow = wave * 16 + rlo;
  int k0l = hf * 16;
  int kk0 = hf * 16;

  for (int k0 = 0; k0 < Kd; k0 += BK) {
    stage_store();
    __syncthreads();
    if (k0 + BK < Kd) stage_load(k0 + BK);  // overlap next global fetch with math
    // A fragment: row = wave*16 + (lane&15); lane half hf holds contiguous k = 16*hf..16*hf+15 (B same)
    bf16x8 alo = *(const bf16x8*)&As[arow][k0l];
    bf16x8 ahi = *(const bf16x8*)&As[arow][k0l + 8];
    bf16x16 afrag = join16(alo, ahi);
    // preload all 8 B fragments, then 8 back-to-back WMMAs
    bf16x16 bfr[8];
#pragma unroll
    for (int t = 0; t < 8; ++t)
      bfr[t] = *(const bf16x16*)&Bst[t * 16 + rlo][kk0];
#pragma unroll
    for (int t = 0; t < 8; ++t) acc[t] = wmma_bf16(afrag, bfr[t], acc[t]);
    __syncthreads();
  }

  // epilogue — C layout: row = i + 8*hf, col = lane&15
#pragma unroll
  for (int t = 0; t < 8; ++t) {
    int col = c0 + t * 16 + rlo;
#pragma unroll
    for (int i = 0; i < 8; ++i) {
      int r = r0 + wave * 16 + hf * 8 + i;
      if (r >= valid) continue;
      float v = acc[t][i];
      if (mode == MODE_STORE) {
        outf[(size_t)r * ldo + col] = v;
      } else if (mode == MODE_ADD) {
        outf[(size_t)r * ldo + col] += v;
      } else if (mode == MODE_SILU_MUL) {
        float sl = v / (1.f + __expf(-v));
        outbf[(size_t)r * ldo + col] = f2bf(sl * aux[(size_t)r * ldo + col]);
      } else { // MODE_SCATTER
        int tr = omap[r];
        outf[(size_t)tr * ldo + col] += oscale[r] * v;
      }
    }
  }
}

// ---------------- RoPE + bf16 repack ----------------
__global__ void tg_rope_q(const float* __restrict__ qf, __bf16* __restrict__ Q) {
  int i = blockIdx.x * 256 + threadIdx.x; // N*H*32
  int j = i & 31;
  int rest = i >> 5;
  int hh = rest % HH; rest /= HH;
  int s = rest % SS;
  int b = rest / SS;
  int n = b * SS + s;
  float xe = qf[(size_t)n * (HH * HDD) + hh * HDD + 2 * j];
  float xo = qf[(size_t)n * (HH * HDD) + hh * HDD + 2 * j + 1];
  float fr = __powf(1.0e6f, -(float)j / 32.f);
  float ang = (float)s * fr;
  float sn, cs;
  __sincosf(ang, &sn, &cs);
  size_t o = ((size_t)(b * HH + hh) * SS + s) * HDD + 2 * j;
  *(unsigned*)(Q + o) = pkbf(xe * cs - xo * sn, xe * sn + xo * cs);
}
__global__ void tg_rope_k(const float* __restrict__ kf, __bf16* __restrict__ Kc) {
  int i = blockIdx.x * 256 + threadIdx.x; // N*KV*32
  int j = i & 31;
  int rest = i >> 5;
  int kv = rest % KVH; rest /= KVH;
  int s = rest % SS;
  int b = rest / SS;
  int n = b * SS + s;
  float xe = kf[(size_t)n * (KVH * HDD) + kv * HDD + 2 * j];
  float xo = kf[(size_t)n * (KVH * HDD) + kv * HDD + 2 * j + 1];
  float fr = __powf(1.0e6f, -(float)j / 32.f);
  float ang = (float)s * fr;
  float sn, cs;
  __sincosf(ang, &sn, &cs);
  size_t o = ((size_t)(b * KVH + kv) * SS + s) * HDD + 2 * j;
  *(unsigned*)(Kc + o) = pkbf(xe * cs - xo * sn, xe * sn + xo * cs);
}
// V repack TRANSPOSED: [b][kv][hd][S] so PV B-fragments are contiguous 32B loads
__global__ void tg_conv_vT(const float* __restrict__ vf, __bf16* __restrict__ Vt) {
  int i = blockIdx.x * 256 + threadIdx.x; // B*KV*HD*S
  int s = i & (SS - 1);
  int rest = i >> 10;
  int d = rest & (HDD - 1);
  rest >>= 6;
  int kv = rest % KVH;
  int b = rest / KVH;
  Vt[(((size_t)(b * KVH + kv)) * HDD + d) * SS + s] =
      f2bf(vf[((size_t)(b * SS + s)) * (KVH * HDD) + kv * HDD + d]);
}

// ---------------- flash attention (one wave / 16 q-rows) ----------------
__global__ __launch_bounds__(32) void tg_attn(const __bf16* __restrict__ Q,
                                              const __bf16* __restrict__ Kc,
                                              const __bf16* __restrict__ Vt,
                                              __bf16* __restrict__ O) {
  __shared__ __bf16 P[16][32];
  int lane = threadIdx.x;
  int rlo = lane & 15, hf = lane >> 4;
  int q0 = blockIdx.x * 16;
  int hh = blockIdx.y;
  int b = blockIdx.z;
  int kvh = hh / REPF;
  const __bf16* Qp = Q + (((size_t)(b * HH + hh)) * SS + q0) * HDD;
  const __bf16* Kp = Kc + ((size_t)(b * KVH + kvh)) * SS * HDD;
  const __bf16* Vp = Vt + ((size_t)(b * KVH + kvh)) * HDD * SS;

  // Q fragments for HD split into two K=32 steps
  bf16x16 aq[2];
  {
    int k0l = hf * 16;
#pragma unroll
    for (int sgl = 0; sgl < 2; ++sgl) {
      bf16x8 lo = *(const bf16x8*)(Qp + (size_t)rlo * HDD + sgl * 32 + k0l);
      bf16x8 hi = *(const bf16x8*)(Qp + (size_t)rlo * HDD + sgl * 32 + k0l + 8);
      aq[sgl] = join16(lo, hi);
    }
  }

  f32x8 oacc[4];
#pragma unroll
  for (int t = 0; t < 4; ++t) oacc[t] = zero8();
  float mr[8], lr[8];
#pragma unroll
  for (int i = 0; i < 8; ++i) { mr[i] = -1e30f; lr[i] = 0.f; }

  int kk0 = hf * 16;
  for (int kb = 0; kb < q0 + 16; kb += 32) {
    // ---- scores: preload 4 K fragments, then 4 WMMAs ----
    bf16x16 bk[4];
#pragma unroll
    for (int sgl = 0; sgl < 2; ++sgl) {
      bk[sgl * 2 + 0] =
          *(const bf16x16*)(Kp + (size_t)(kb + rlo) * HDD + sgl * 32 + kk0);
      bk[sgl * 2 + 1] =
          *(const bf16x16*)(Kp + (size_t)(kb + 16 + rlo) * HDD + sgl * 32 + kk0);
    }
    f32x8 c0 = zero8(), c1 = zero8();
    c0 = wmma_bf16(aq[0], bk[0], c0);
    c1 = wmma_bf16(aq[0], bk[1], c1);
    c0 = wmma_bf16(aq[1], bk[2], c0);
    c1 = wmma_bf16(aq[1], bk[3], c1);

    // ---- online softmax (row stats live in half-wave lane groups) ----
    float p0a[8], p1a[8], sc[8];
#pragma unroll
    for (int i = 0; i < 8; ++i) {
      int qrow = q0 + hf * 8 + i;
      int key0 = kb + rlo, key1 = kb + 16 + rlo;
      float v0 = c0[i] * 0.125f + (key0 > qrow ? -1e30f : 0.f);
      float v1 = c1[i] * 0.125f + (key1 > qrow ? -1e30f : 0.f);
      float bm = fmaxf(v0, v1);
#pragma unroll
      for (int m = 1; m < 16; m <<= 1) bm = fmaxf(bm, __shfl_xor(bm, m, 32));
      float nm = fmaxf(mr[i], bm);
      float scale = __expf(mr[i] - nm);
      float p0 = __expf(v0 - nm), p1 = __expf(v1 - nm);
      float ps = p0 + p1;
#pragma unroll
      for (int m = 1; m < 16; m <<= 1) ps += __shfl_xor(ps, m, 32);
      lr[i] = lr[i] * scale + ps;
      mr[i] = nm;
      sc[i] = scale;
      p0a[i] = p0;
      p1a[i] = p1;
    }
#pragma unroll
    for (int t = 0; t < 4; ++t)
#pragma unroll
      for (int i = 0; i < 8; ++i) oacc[t][i] *= sc[i];

    // ---- stage P through LDS: C-layout -> A-layout ----
#pragma unroll
    for (int i = 0; i < 8; ++i) {
      int r = hf * 8 + i;
      P[r][rlo] = f2bf(p0a[i]);
      P[r][16 + rlo] = f2bf(p1a[i]);
    }
    __syncthreads();
    bf16x16 pa;
    {
      int k0l = hf * 16;
      bf16x8 lo = *(const bf16x8*)&P[rlo][k0l];
      bf16x8 hi = *(const bf16x8*)&P[rlo][k0l + 8];
      pa = join16(lo, hi);
    }
    // ---- PV: transposed V gives contiguous B fragments ----
    bf16x16 bv[4];
#pragma unroll
    for (int t = 0; t < 4; ++t)
      bv[t] = *(const bf16x16*)(Vp + (size_t)(t * 16 + rlo) * SS + kb + kk0);
#pragma unroll
    for (int t = 0; t < 4; ++t) oacc[t] = wmma_bf16(pa, bv[t], oacc[t]);
    __syncthreads();
  }

#pragma unroll
  for (int i = 0; i < 8; ++i) {
    float inv = 1.f / lr[i];
#pragma unroll
    for (int t = 0; t < 4; ++t) oacc[t][i] *= inv;
  }
#pragma unroll
  for (int t = 0; t < 4; ++t)
#pragma unroll
    for (int i = 0; i < 8; ++i) {
      size_t row = (size_t)b * SS + q0 + hf * 8 + i;
      O[row * (HH * HDD) + hh * HDD + t * 16 + rlo] = f2bf(oacc[t][i]);
    }
}

// ---------------- gate: logits, top-2 softmax, bucket tokens ----------------
__global__ void tg_zero_cnt(int* cnt) {
  if (threadIdx.x < EE) cnt[threadIdx.x] = 0;
}
__global__ __launch_bounds__(256) void tg_gate(const __bf16* __restrict__ xn,
                                               const float* __restrict__ gw,
                                               int* __restrict__ cnt,
                                               int* __restrict__ tok,
                                               float* __restrict__ wts) {
  __shared__ float lg[EE];
  int n = blockIdx.x, tid = threadIdx.x;
  int e = tid >> 5, lane = tid & 31;
  const __bf16* xr = xn + (size_t)n * DD;
  float s = 0.f;
  for (int d = lane; d < DD; d += 32) s += bf2f(xr[d]) * gw[(size_t)d * EE + e];
#pragma unroll
  for (int m = 1; m < 32; m <<= 1) s += __shfl_xor(s, m, 32);
  if (lane == 0) lg[e] = s;
  __syncthreads();
  if (tid == 0) {
    int i0 = 0;
    float l0 = lg[0];
    for (int i = 1; i < EE; ++i)
      if (lg[i] > l0) { l0 = lg[i]; i0 = i; }
    int i1 = -1;
    float l1 = -3e38f;
    for (int i = 0; i < EE; ++i)
      if (i != i0 && lg[i] > l1) { l1 = lg[i]; i1 = i; }
    float e1 = __expf(l1 - l0);
    float inv = 1.f / (1.f + e1);
    int s0 = atomicAdd(&cnt[i0], 1);
    tok[i0 * NTOK + s0] = n;
    wts[i0 * NTOK + s0] = inv;
    int s1 = atomicAdd(&cnt[i1], 1);
    tok[i1 * NTOK + s1] = n;
    wts[i1 * NTOK + s1] = e1 * inv;
  }
}

// ---------------- host launcher ----------------
static void run_gemm(hipStream_t st, const __bf16* A, int lda, const float* W,
                     int ldb, int M, int Kd, int Nd, const int* amap,
                     const int* cntPtr, int cntIdx, int mode, int ldo, float* outf,
                     __bf16* outbf, const float* aux, const int* omap,
                     const float* oscale) {
  dim3 g(Nd / BN, M / BM);
  tg_gemm<<<g, 256, 0, st>>>(A, lda, W, ldb, M, Kd, amap, cntPtr, cntIdx, mode,
                             ldo, outf, outbf, aux, omap, oscale);
}

extern "C" void kernel_launch(void* const* d_in, const int* in_sizes, int n_in,
                              void* d_out, int out_size, void* d_ws, size_t ws_size,
                              hipStream_t stream) {
  (void)in_sizes; (void)n_in; (void)out_size; (void)ws_size;
  const int*   ids          = (const int*)d_in[0];
  const float* tok_emb      = (const float*)d_in[2];
  const float* attn_norm_w  = (const float*)d_in[3];
  const float* ffn_norm_w   = (const float*)d_in[4];
  const float* wq           = (const float*)d_in[5];
  const float* wk           = (const float*)d_in[6];
  const float* wv           = (const float*)d_in[7];
  const float* wo           = (const float*)d_in[8];
  const float* gate_w       = (const float*)d_in[9];
  const float* w1           = (const float*)d_in[10];
  const float* w2           = (const float*)d_in[11];
  const float* w3           = (const float*)d_in[12];
  const float* final_norm_w = (const float*)d_in[13];
  const float* out_w        = (const float*)d_in[14];
  float* out = (float*)d_out;

  char* ws = (char*)d_ws;
  size_t off = 0;
  auto take = [&](size_t bytes) -> void* {
    void* p = ws + off;
    off = (off + bytes + 255) & ~(size_t)255;
    return p;
  };
  float*  h   = (float*)take((size_t)NTOK * DD * 4);
  __bf16* xn  = (__bf16*)take((size_t)NTOK * DD * 2);
  float*  qf  = (float*)take((size_t)NTOK * HH * HDD * 4);
  float*  kf  = (float*)take((size_t)NTOK * KVH * HDD * 4);
  float*  vf  = (float*)take((size_t)NTOK * KVH * HDD * 4);
  __bf16* Qb  = (__bf16*)take((size_t)BB * HH * SS * HDD * 2);
  __bf16* Kb  = (__bf16*)take((size_t)BB * KVH * SS * HDD * 2);
  __bf16* Vb  = (__bf16*)take((size_t)BB * KVH * SS * HDD * 2);
  __bf16* Ob  = (__bf16*)take((size_t)NTOK * HH * HDD * 2);
  float*  hb3 = (float*)take((size_t)NTOK * FF * 4);
  __bf16* h1b = (__bf16*)take((size_t)NTOK * FF * 2);
  int*    cnt = (int*)take(EE * 4);
  int*    tok = (int*)take((size_t)EE * NTOK * 4);
  float*  wts = (float*)take((size_t)EE * NTOK * 4);

  tg_embed<<<(NTOK * DD) / 256, 256, 0, stream>>>(ids, tok_emb, h);

  for (int l = 0; l < LL; ++l) {
    const float* wq_l = wq + (size_t)l * DD * (HH * HDD);
    const float* wk_l = wk + (size_t)l * DD * (KVH * HDD);
    const float* wv_l = wv + (size_t)l * DD * (KVH * HDD);
    const float* wo_l = wo + (size_t)l * (HH * HDD) * DD;
    const float* gate_l = gate_w + (size_t)l * DD * EE;

    // --- attention block ---
    tg_rmsnorm<<<NTOK, 256, 0, stream>>>(h, attn_norm_w + (size_t)l * DD, xn);
    run_gemm(stream, xn, DD, wq_l, HH * HDD, NTOK, DD, HH * HDD, nullptr, nullptr,
             0, MODE_STORE, HH * HDD, qf, nullptr, nullptr, nullptr, nullptr);
    run_gemm(stream, xn, DD, wk_l, KVH * HDD, NTOK, DD, KVH * HDD, nullptr,
             nullptr, 0, MODE_STORE, KVH * HDD, kf, nullptr, nullptr, nullptr,
             nullptr);
    run_gemm(stream, xn, DD, wv_l, KVH * HDD, NTOK, DD, KVH * HDD, nullptr,
             nullptr, 0, MODE_STORE, KVH * HDD, vf, nullptr, nullptr, nullptr,
             nullptr);
    tg_rope_q<<<(NTOK * HH * (HDD / 2)) / 256, 256, 0, stream>>>(qf, Qb);
    tg_rope_k<<<(NTOK * KVH * (HDD / 2)) / 256, 256, 0, stream>>>(kf, Kb);
    tg_conv_vT<<<(NTOK * KVH * HDD) / 256, 256, 0, stream>>>(vf, Vb);
    tg_attn<<<dim3(SS / 16, HH, BB), 32, 0, stream>>>(Qb, Kb, Vb, Ob);
    run_gemm(stream, Ob, HH * HDD, wo_l, DD, NTOK, HH * HDD, DD, nullptr, nullptr,
             0, MODE_ADD, DD, h, nullptr, nullptr, nullptr, nullptr);

    // --- MoE block ---
    tg_rmsnorm<<<NTOK, 256, 0, stream>>>(h, ffn_norm_w + (size_t)l * DD, xn);
    tg_zero_cnt<<<1, 32, 0, stream>>>(cnt);
    tg_gate<<<NTOK, 256, 0, stream>>>(xn, gate_l, cnt, tok, wts);
    for (int e = 0; e < EE; ++e) {
      const float* w1_le = w1 + ((size_t)l * EE + e) * DD * FF;
      const float* w2_le = w2 + ((size_t)l * EE + e) * FF * DD;
      const float* w3_le = w3 + ((size_t)l * EE + e) * DD * FF;
      const int* tok_e = tok + (size_t)e * NTOK;
      const float* wt_e = wts + (size_t)e * NTOK;
      // h3 = xg @ w3   (gathered rows, stored f32 by slot)
      run_gemm(stream, xn, DD, w3_le, FF, NTOK, DD, FF, tok_e, cnt, e, MODE_STORE,
               FF, hb3, nullptr, nullptr, nullptr, nullptr);
      // h1 = silu(xg @ w1) * h3  -> bf16 by slot
      run_gemm(stream, xn, DD, w1_le, FF, NTOK, DD, FF, tok_e, cnt, e,
               MODE_SILU_MUL, FF, nullptr, h1b, hb3, nullptr, nullptr);
      // h += wt * (h1 @ w2)  (scatter by token)
      run_gemm(stream, h1b, FF, w2_le, DD, NTOK, FF, DD, nullptr, cnt, e,
               MODE_SCATTER, DD, h, nullptr, nullptr, tok_e, wt_e);
    }
  }

  // --- final norm + vocab projection ---
  tg_rmsnorm<<<NTOK, 256, 0, stream>>>(h, final_norm_w, xn);
  run_gemm(stream, xn, DD, out_w, VV, NTOK, DD, VV, nullptr, nullptr, 0,
           MODE_STORE, VV, out, nullptr, nullptr, nullptr, nullptr);
}
